// Attention_3212635538082
// MI455X (gfx1250) — matrix operand order repaired and checked by reading, not compiled, not run on hardware
//
#include <hip/hip_runtime.h>
#include <stdint.h>

// CDNA5 / gfx1250 flash-attention (causal, key==value), fp32 IO.
// f16 WMMA core + TDM (tensor_load_to_lds) double-buffered K/V staging.

typedef __attribute__((ext_vector_type(16))) _Float16 v16h;
typedef __attribute__((ext_vector_type(8)))  _Float16 v8h;
typedef __attribute__((ext_vector_type(8)))  float    v8f;
typedef __attribute__((ext_vector_type(4)))  unsigned int u32x4;
typedef __attribute__((ext_vector_type(8)))  unsigned int u32x8;

#define WMMA_F16(A,B,C) __builtin_amdgcn_wmma_f32_16x16x32_f16( \
    false, (A), false, (B), (short)0, (C), false, false)

constexpr int BB = 8;       // batch
constexpr int TQ = 2048;    // query length
constexpr int TV = 2048;    // key/value length
constexpr int DH = 64;      // head dim
constexpr int WAVES = 8;    // waves per block (wave32)
constexpr int QROWS_PER_BLOCK = WAVES * 16;
constexpr int KT = 32;      // keys per iteration
constexpr float NEG_BIG = 1.0e9f;

union H2U { unsigned int u; _Float16 h[2]; };

// ---- TDM: DMA one 32x64 fp32 tile (rows of K/V) from global into LDS. ----
// D# group0/group1 per CDNA5 ISA (2-D tensor, 2-group form).
__device__ __forceinline__ void tdm_load_tile(const float* gsrc, void* ldst) {
    unsigned long long ga = (unsigned long long)(uintptr_t)gsrc;
    unsigned int lds_off  = (unsigned int)(uintptr_t)ldst;   // low 32b of generic = LDS byte offset
    u32x4 g0;
    g0[0] = 1u;                                              // count=1, user D#
    g0[1] = lds_off;                                         // lds_addr
    g0[2] = (unsigned int)(ga & 0xFFFFFFFFu);                // global_addr[31:0]
    g0[3] = (unsigned int)((ga >> 32) & 0x01FFFFFFu)         // global_addr[56:32]
          | (2u << 30);                                      // type = 2 ("image")
    u32x8 g1;
    g1[0] = (2u << 16);              // workgroup_mask=0, data_size=2 (4 bytes)
    g1[1] = (unsigned int)DH << 16;  // tensor_dim0[15:0] = 64
    g1[2] = ((unsigned int)TV) << 16;// tensor_dim0[31:16]=0 | tensor_dim1[15:0] = 2048
    g1[3] = ((unsigned int)DH) << 16;// tensor_dim1[31:16]=0 | tile_dim0 = 64
    g1[4] = (unsigned int)KT;        // tile_dim1 = 32, tile_dim2 = 0
    g1[5] = (unsigned int)DH;        // tensor_dim0_stride[31:0] = 64
    g1[6] = 0u;                      // stride0[47:32]=0, stride1[15:0]=0
    g1[7] = 0u;
    asm volatile("tensor_load_to_lds %0, %1" :: "s"(g0), "s"(g1) : "memory");
}

__global__ __launch_bounds__(256)
void fa_wmma_kernel(const float* __restrict__ Q,
                    const float* __restrict__ V,
                    const float* __restrict__ scale,
                    const uint8_t* __restrict__ qmask,
                    const uint8_t* __restrict__ vmask,
                    float* __restrict__ Out)
{
    // Double-buffered raw fp32 tile (TDM target) and f16 tiles in both majors.
    __shared__ __align__(16) float     rawV[2][KT][DH];        // 16 KB
    __shared__ __align__(16) _Float16  VhR [2][KT][DH + 8];    // row-major  (pad 8)
    __shared__ __align__(16) _Float16  VhT [2][DH][KT + 8];    // d-major    (pad 8)
    __shared__ __align__(16) _Float16  ldsP[WAVES][16][34];    // P transpose scratch

    const int tid  = threadIdx.x;
    const int lane = tid & 31;
    const int wave = tid >> 5;
    const int h    = lane >> 4;     // lane half-group (0/1)
    const int mrow = lane & 15;     // M (or N) index within 16

    const int blocksPerBatch = TQ / QROWS_PER_BLOCK;     // 16
    const int b     = blockIdx.x / blocksPerBatch;
    const int qBlk  = blockIdx.x % blocksPerBatch;
    const int qBase = qBlk * QROWS_PER_BLOCK + wave * 16;

    const float sc = scale[0];
    const float*   Qb = Q + (size_t)b * TQ * DH;
    const float*   Vb = V + (size_t)b * TV * DH;
    const uint8_t* vm = vmask + (size_t)b * TV;
    const uint8_t* qm = qmask + (size_t)b * TQ;

    // ---------------- Q tile (16x64) -> two f16 A fragments -----------------
    v16h aQ[2];
    {
        const float* qrow = Qb + (size_t)(qBase + mrow) * DH;
        #pragma unroll
        for (int f = 0; f < 2; ++f) {
            #pragma unroll
            for (int v = 0; v < 8; ++v) {
                const int d = f * 32 + (v >> 2) * 16 + h * 8 + (v & 3) * 2;
                const float2 q2 = *(const float2*)(qrow + d);
                aQ[f][2 * v]     = (_Float16)q2.x;
                aQ[f][2 * v + 1] = (_Float16)q2.y;
            }
        }
    }

    v8f O[4];
    #pragma unroll
    for (int i = 0; i < 4; ++i) O[i] = (v8f){0,0,0,0,0,0,0,0};
    float mrun[8], lrun[8];
    #pragma unroll
    for (int j = 0; j < 8; ++j) { mrun[j] = -INFINITY; lrun[j] = 0.0f; }

    const int lastK = qBase + 15;        // furthest key this wave can see
    const int nT    = 4 * (qBlk + 1);    // block-uniform tile count (causal)

    const v8f zero8 = (v8f){0,0,0,0,0,0,0,0};

    // Prologue: DMA tile 0.
    if (wave == 0) tdm_load_tile(Vb, &rawV[0][0][0]);

    // Convert-pass indices: each thread owns an 8-element row segment.
    const int crow = tid >> 3;          // 0..31
    const int cd0  = (tid & 7) * 8;     // 0..56

    for (int t = 0; t < nT; ++t) {
        const int kBase = t * KT;
        const int buf   = t & 1;

        // ---- wait for TDM tile t, then shared fp32 -> f16 convert pass ----
        if (wave == 0) __builtin_amdgcn_s_wait_tensorcnt(0);
        __syncthreads();

        {
            const float4 x0 = *(const float4*)&rawV[buf][crow][cd0];
            const float4 x1 = *(const float4*)&rawV[buf][crow][cd0 + 4];
            v8h pk;
            pk[0] = (_Float16)x0.x; pk[1] = (_Float16)x0.y;
            pk[2] = (_Float16)x0.z; pk[3] = (_Float16)x0.w;
            pk[4] = (_Float16)x1.x; pk[5] = (_Float16)x1.y;
            pk[6] = (_Float16)x1.z; pk[7] = (_Float16)x1.w;
            *(v8h*)&VhR[buf][crow][cd0] = pk;           // row-major f16
            #pragma unroll
            for (int e = 0; e < 8; ++e)                 // transposed f16
                VhT[buf][cd0 + e][crow] = pk[e];
        }
        __syncthreads();

        // ---- kick off DMA for tile t+1 (overlaps with compute below) ----
        if (wave == 0 && (t + 1) < nT)
            tdm_load_tile(Vb + (size_t)(t + 1) * KT * DH, &rawV[(t + 1) & 1][0][0]);

        if (kBase <= lastK) {
            // ---- bK fragments from row-major f16 LDS ----
            v16h bK0[2], bK1[2];
            #pragma unroll
            for (int f = 0; f < 2; ++f) {
                const int d0 = f * 32 + h * 8;
                const v8h a0 = *(const v8h*)&VhR[buf][mrow][d0];
                const v8h a1 = *(const v8h*)&VhR[buf][mrow][d0 + 16];
                const v8h b0 = *(const v8h*)&VhR[buf][16 + mrow][d0];
                const v8h b1 = *(const v8h*)&VhR[buf][16 + mrow][d0 + 16];
                #pragma unroll
                for (int i = 0; i < 8; ++i) {
                    bK0[f][i] = a0[i]; bK0[f][8 + i] = a1[i];
                    bK1[f][i] = b0[i]; bK1[f][8 + i] = b1[i];
                }
            }

            v8f s0 = WMMA_F16(aQ[0], bK0[0], zero8);
            s0     = WMMA_F16(aQ[1], bK0[1], s0);
            v8f s1 = WMMA_F16(aQ[0], bK1[0], zero8);
            s1     = WMMA_F16(aQ[1], bK1[1], s1);

            // ---- scale + mask + online softmax ----
            const int col0 = kBase + mrow;
            const int col1 = kBase + 16 + mrow;
            const float pen0 = vm[col0] ? 0.0f : NEG_BIG;
            const float pen1 = vm[col1] ? 0.0f : NEG_BIG;

            #pragma unroll
            for (int j = 0; j < 8; ++j) {
                const int qg = qBase + j + h * 8;
                float t0 = s0[j] * sc - ((col0 <= qg) ? pen0 : NEG_BIG);
                float t1 = s1[j] * sc - ((col1 <= qg) ? pen1 : NEG_BIG);

                float mx = fmaxf(t0, t1);
                mx = fmaxf(mx, __shfl_xor(mx, 1, 32));
                mx = fmaxf(mx, __shfl_xor(mx, 2, 32));
                mx = fmaxf(mx, __shfl_xor(mx, 4, 32));
                mx = fmaxf(mx, __shfl_xor(mx, 8, 32));

                const float mnew  = fmaxf(mrun[j], mx);
                const float alpha = __expf(mrun[j] - mnew);
                const float p0 = __expf(t0 - mnew);
                const float p1 = __expf(t1 - mnew);

                float rs = p0 + p1;
                rs += __shfl_xor(rs, 1, 32);
                rs += __shfl_xor(rs, 2, 32);
                rs += __shfl_xor(rs, 4, 32);
                rs += __shfl_xor(rs, 8, 32);

                lrun[j] = lrun[j] * alpha + rs;
                mrun[j] = mnew;
                O[0][j] *= alpha; O[1][j] *= alpha;
                O[2][j] *= alpha; O[3][j] *= alpha;

                ldsP[wave][j + h * 8][mrow]      = (_Float16)p0;
                ldsP[wave][j + h * 8][16 + mrow] = (_Float16)p1;
            }

            // Per-wave LDS handoff (in-order per wave); stop reordering.
            asm volatile("" ::: "memory");
            __builtin_amdgcn_wave_barrier();

            v16h aP;
            #pragma unroll
            for (int v = 0; v < 8; ++v) {
                const int k0 = (v >> 2) * 16 + h * 8 + (v & 3) * 2;   // even
                H2U cv;
                cv.u = *(const unsigned int*)&ldsP[wave][mrow][k0];
                aP[2 * v]     = cv.h[0];
                aP[2 * v + 1] = cv.h[1];
            }
            asm volatile("" ::: "memory");
            __builtin_amdgcn_wave_barrier();

            // ---- O(16x64) += P(16x32) * V(32x64), bV from transposed LDS ----
            #pragma unroll
            for (int f2 = 0; f2 < 4; ++f2) {
                const int d = f2 * 16 + mrow;
                const v8h c0 = *(const v8h*)&VhT[buf][d][h * 8];
                const v8h c1 = *(const v8h*)&VhT[buf][d][h * 8 + 16];
                v16h bV;
                #pragma unroll
                for (int i = 0; i < 8; ++i) { bV[i] = c0[i]; bV[8 + i] = c1[i]; }
                O[f2] = WMMA_F16(aP, bV, O[f2]);
            }
        }
    }

    // ---------------- finalize: /l, q_mask, store fp32 ----------------
    #pragma unroll
    for (int j = 0; j < 8; ++j) {
        const int qg = qBase + j + h * 8;
        const float inv = (lrun[j] > 0.0f) ? (1.0f / lrun[j]) : 0.0f;
        const float s   = inv * (qm[qg] ? 1.0f : 0.0f);
        float* orow = Out + ((size_t)b * TQ + qg) * DH;
        #pragma unroll
        for (int f2 = 0; f2 < 4; ++f2)
            orow[f2 * 16 + mrow] = O[f2][j] * s;
    }
}

extern "C" void kernel_launch(void* const* d_in, const int* in_sizes, int n_in,
                              void* d_out, int out_size, void* d_ws, size_t ws_size,
                              hipStream_t stream) {
    const float*   Q     = (const float*)d_in[0];
    const float*   V     = (const float*)d_in[1];
    const float*   scale = (const float*)d_in[2];
    const uint8_t* qmask = (const uint8_t*)d_in[3];
    const uint8_t* vmask = (const uint8_t*)d_in[4];
    float* out = (float*)d_out;

    dim3 grid(BB * (TQ / QROWS_PER_BLOCK));   // 8 * 16 = 128 blocks
    dim3 block(32 * WAVES);                   // 256 threads = 8 wave32
    fa_wmma_kernel<<<grid, block, 0, stream>>>(Q, V, scale, qmask, vmask, out);
}
